// Attention_cross_35150012351238
// MI455X (gfx1250) — matrix operand order repaired and checked by reading, not compiled, not run on hardware
//
#include <hip/hip_runtime.h>
#include <hip/hip_bf16.h>

typedef __bf16 bf16;
typedef __attribute__((ext_vector_type(16))) __bf16 v16bf;
typedef __attribute__((ext_vector_type(8)))  __bf16 v8bf;
typedef __attribute__((ext_vector_type(2)))  __bf16 bf162;
typedef __attribute__((ext_vector_type(8)))  float  v8f;

// ---------------------------------------------------------------------------
// CDNA5 async copy: global (16B/lane) -> LDS, tracked on ASYNCcnt.
// LDS address = low 32 bits of the generic pointer (LDS aperture: addr[31:0]).
// ---------------------------------------------------------------------------
__device__ __forceinline__ void async_copy_b128(void* lds_dst, const void* gsrc) {
  unsigned lds = (unsigned)(unsigned long long)lds_dst;
  unsigned long long g = (unsigned long long)gsrc;
  asm volatile("global_load_async_to_lds_b128 %0, %1, off"
               :: "v"(lds), "v"(g) : "memory");
}
__device__ __forceinline__ void wait_async0() {
  asm volatile("s_wait_asynccnt 0" ::: "memory");
}

// ---------------------------------------------------------------------------
// WMMA helpers (wave32; A/B use contiguous k = 16*(lane/16) + i, same order in both)
// ---------------------------------------------------------------------------
// A-matrix 16x32 bf16 from row-major source (leading dim ld).
__device__ __forceinline__ v16bf load_A16x32(const bf16* base, int ld) {
  const int lane = threadIdx.x & 31;
  const int row  = lane & 15;
  const int kb   = (lane & 16) ? 16 : 0;
  const bf16* p = base + row * ld + kb;
  v16bf a;
#pragma unroll
  for (int i = 0; i < 16; ++i) a[i] = p[i];
  return a;
}
// B-matrix 32(K)x16(N) bf16 from a TRANSPOSED store Bt[n][k] (leading dim ldt).
__device__ __forceinline__ v16bf load_B32x16(const bf16* baseT, int ldt) {
  const int lane = threadIdx.x & 31;
  const int n  = lane & 15;
  const int kb = (lane & 16) ? 16 : 0;
  const bf16* p = baseT + n * ldt + kb;
  v16bf b;
#pragma unroll
  for (int i = 0; i < 16; ++i) b[i] = p[i];
  return b;
}
__device__ __forceinline__ v8f wmma_bf16(v16bf a, v16bf b, v8f c) {
  return __builtin_amdgcn_wmma_f32_16x16x32_bf16(false, a, false, b, (short)0, c,
                                                 false, false);
}

// 16-lane row reductions matching the C/D layout (cols on lanes 0..15 / 16..31)
__device__ __forceinline__ float rmax16(float v) {
  v = fmaxf(v, __shfl_xor(v, 1));
  v = fmaxf(v, __shfl_xor(v, 2));
  v = fmaxf(v, __shfl_xor(v, 4));
  v = fmaxf(v, __shfl_xor(v, 8));
  return v;
}
__device__ __forceinline__ float rsum16(float v) {
  v += __shfl_xor(v, 1);
  v += __shfl_xor(v, 2);
  v += __shfl_xor(v, 4);
  v += __shfl_xor(v, 8);
  return v;
}

// ---------------------------------------------------------------------------
// GEMM: C[M,Nc] = map(A)[M,K] @ Bw[K,Nc] (+bias). 128x128 block tile, 8 waves,
// each wave 64x32 (8 WMMAs / 32-K step). Row mapping (RPB = 1024, compile-time)
// handles x[:, :N]/x[:, N:] slicing and the concat output placement.
// ---------------------------------------------------------------------------
template <int RPB, typename AT, typename OT>
__global__ __launch_bounds__(256)
void gemm_bf16_wmma(const AT* __restrict__ A, const float* __restrict__ Bw,
                    const float* __restrict__ bias, OT* __restrict__ Cout,
                    int M, int Nc, int K,
                    int aBatchStride, int aRowOff,
                    int cBatchStride, int cRowOff) {
  __shared__ bf16 As[128][32];    // [m][k]
  __shared__ bf16 Bt[128][32];    // [n][k] transposed for B-layout gathers
  const int tid  = threadIdx.x;
  const int wave = tid >> 5, lane = tid & 31;
  const int wm = wave >> 2;            // 0..1 : 64-row strip
  const int wn = wave & 3;             // 0..3 : 32-col strip
  const int m0 = blockIdx.x * 128;
  const int n0 = blockIdx.y * 128;

  v8f acc[4][2];
#pragma unroll
  for (int i = 0; i < 4; ++i)
#pragma unroll
    for (int j = 0; j < 2; ++j)
#pragma unroll
      for (int r = 0; r < 8; ++r) acc[i][j][r] = 0.f;

  // A staging: 2 threads/row, 16 contiguous elems each
  const int arow = tid >> 1;
  const int aseg = (tid & 1) * 16;
  const int grow = m0 + arow;
  const long aRowG = (long)(grow >> 10) * aBatchStride + (grow & (RPB - 1)) + aRowOff;
  const AT* aSrcRow = A + aRowG * (long)K;
  // B staging: thread t -> k-pair p = t>>4, 8 columns (t&15)*8 (packed b32 stores)
  const int bp  = (tid >> 4) * 2;
  const int bns = (tid & 15) * 8;

  for (int k0 = 0; k0 < K; k0 += 32) {
    __syncthreads();
    // ---- stage A (async when already bf16, convert when f32) ----
    if constexpr (__is_same(AT, bf16)) {
      async_copy_b128(&As[arow][aseg], aSrcRow + k0 + aseg);
      async_copy_b128(&As[arow][aseg + 8], aSrcRow + k0 + aseg + 8);
    } else {
#pragma unroll
      for (int half = 0; half < 2; ++half) {
        v8bf pk;
#pragma unroll
        for (int i = 0; i < 8; ++i)
          pk[i] = (bf16)aSrcRow[k0 + aseg + half * 8 + i];
        *(v8bf*)&As[arow][aseg + half * 8] = pk;
      }
    }
    // ---- stage B transposed, two k-rows packed per b32 store ----
    {
      const float* blo = Bw + (long)(k0 + bp) * Nc + n0 + bns;
      const float* bhi = blo + Nc;
#pragma unroll
      for (int i = 0; i < 8; ++i) {
        bf162 pr; pr[0] = (bf16)blo[i]; pr[1] = (bf16)bhi[i];
        *(bf162*)&Bt[bns + i][bp] = pr;
      }
    }
    if constexpr (__is_same(AT, bf16)) wait_async0();
    __syncthreads();

    v16bf a0 = load_A16x32(&As[wm * 64][0], 32);
    v16bf a1 = load_A16x32(&As[wm * 64 + 16][0], 32);
    v16bf a2 = load_A16x32(&As[wm * 64 + 32][0], 32);
    v16bf a3 = load_A16x32(&As[wm * 64 + 48][0], 32);
    v16bf b0 = load_B32x16(&Bt[wn * 32][0], 32);
    v16bf b1 = load_B32x16(&Bt[wn * 32 + 16][0], 32);
    acc[0][0] = wmma_bf16(a0, b0, acc[0][0]);
    acc[0][1] = wmma_bf16(a0, b1, acc[0][1]);
    acc[1][0] = wmma_bf16(a1, b0, acc[1][0]);
    acc[1][1] = wmma_bf16(a1, b1, acc[1][1]);
    acc[2][0] = wmma_bf16(a2, b0, acc[2][0]);
    acc[2][1] = wmma_bf16(a2, b1, acc[2][1]);
    acc[3][0] = wmma_bf16(a3, b0, acc[3][0]);
    acc[3][1] = wmma_bf16(a3, b1, acc[3][1]);
  }

  const int hi = (lane & 16) ? 8 : 0;
  const int cn = lane & 15;
#pragma unroll
  for (int i = 0; i < 4; ++i)
#pragma unroll
    for (int j = 0; j < 2; ++j) {
      const int colG = n0 + wn * 32 + j * 16 + cn;
      const float bv = bias ? bias[colG] : 0.f;
#pragma unroll
      for (int r = 0; r < 8; ++r) {
        const int rowL = m0 + wm * 64 + i * 16 + hi + r;
        const long rowG = (long)(rowL >> 10) * cBatchStride +
                          (rowL & (RPB - 1)) + cRowOff;
        Cout[rowG * (long)Nc + colG] = (OT)(acc[i][j][r] + bv);
      }
    }
}

// ---------------------------------------------------------------------------
// Flash attention: softmax(Q K^T * scale) @ V per (b, h). 8 waves x 16 query
// rows; 32-key tiles streamed through LDS (K via async-to-LDS, V transposed).
// ---------------------------------------------------------------------------
template <typename OT>
__global__ __launch_bounds__(256)
void flash_attn_wmma(const bf16* __restrict__ Qp, int qLd,
                     const bf16* __restrict__ Kp, int kLd,
                     const bf16* __restrict__ Vp, int vLd,
                     OT* __restrict__ Op, int oBatchStride, int oRowOff,
                     int Nq, int Nk, int Cc, float scale_log2e) {
  __shared__ bf16 Ks[32][64];       // [key][d] row-major (score-B gathers)
  __shared__ bf16 Vt[64][32];       // [d][key] transposed (PV-B gathers)
  __shared__ bf16 Ps[8][16][32];    // per-wave P tile (C-layout -> A-layout)
  const int tid = threadIdx.x, wave = tid >> 5, lane = tid & 31;
  const int h = blockIdx.y, b = blockIdx.z;
  const int n0 = blockIdx.x * 128 + wave * 16;
  const int hc = h * 64;

  const bf16* qbase = Qp + ((long)(b * Nq + n0)) * qLd + hc;
  v16bf qa0 = load_A16x32(qbase, qLd);
  v16bf qa1 = load_A16x32(qbase + 32, qLd);

  v8f o[4];
  float m_i[8], l_i[8];
#pragma unroll
  for (int t = 0; t < 4; ++t)
#pragma unroll
    for (int r = 0; r < 8; ++r) o[t][r] = 0.f;
#pragma unroll
  for (int r = 0; r < 8; ++r) { m_i[r] = -1e30f; l_i[r] = 0.f; }

  // K staging: 16B per thread (async). V staging: key-pairs packed as b32.
  const int skr  = tid >> 3;        // 0..31
  const int sseg = (tid & 7) * 8;   // 8 contiguous d-elems (16B)
  const int vkp  = (tid >> 4) * 2;  // key pair 0,2,..,30
  const int vds  = (tid & 15) * 4;  // 4 d-elems

  for (int j0 = 0; j0 < Nk; j0 += 32) {
    __syncthreads();
    async_copy_b128(&Ks[skr][sseg],
                    Kp + ((long)(b * Nk + j0 + skr)) * kLd + hc + sseg);
    {
      const bf16* vlo = Vp + ((long)(b * Nk + j0 + vkp)) * vLd + hc + vds;
      const bf16* vhi = vlo + vLd;
#pragma unroll
      for (int i = 0; i < 4; ++i) {
        bf162 pr; pr[0] = vlo[i]; pr[1] = vhi[i];
        *(bf162*)&Vt[vds + i][vkp] = pr;
      }
    }
    wait_async0();
    __syncthreads();

    // S = Q @ K^T : two 16x16 score tiles (keys 0..15 / 16..31)
    v8f s0, s1;
#pragma unroll
    for (int r = 0; r < 8; ++r) { s0[r] = 0.f; s1[r] = 0.f; }
    s0 = wmma_bf16(qa0, load_B32x16(&Ks[0][0], 64), s0);
    s0 = wmma_bf16(qa1, load_B32x16(&Ks[0][32], 64), s0);
    s1 = wmma_bf16(qa0, load_B32x16(&Ks[16][0], 64), s1);
    s1 = wmma_bf16(qa1, load_B32x16(&Ks[16][32], 64), s1);

    // Online softmax in the exp2 domain
    float alpha[8], p0[8], p1[8];
#pragma unroll
    for (int r = 0; r < 8; ++r) {
      s0[r] *= scale_log2e; s1[r] *= scale_log2e;
      float mx = rmax16(fmaxf(s0[r], s1[r]));
      float mn = fmaxf(m_i[r], mx);
      alpha[r] = exp2f(m_i[r] - mn);
      m_i[r] = mn;
    }
#pragma unroll
    for (int r = 0; r < 8; ++r) {
      p0[r] = exp2f(s0[r] - m_i[r]);
      p1[r] = exp2f(s1[r] - m_i[r]);
      l_i[r] = l_i[r] * alpha[r] + rsum16(p0[r] + p1[r]);
    }

    // P: C/D layout -> per-wave LDS -> A layout (16x32)
    const int hi = (lane & 16) ? 8 : 0;
    const int cn = lane & 15;
#pragma unroll
    for (int r = 0; r < 8; ++r) {
      Ps[wave][hi + r][cn]      = (bf16)p0[r];
      Ps[wave][hi + r][16 + cn] = (bf16)p1[r];
    }
    asm volatile("s_wait_dscnt 0" ::: "memory");  // same-wave LDS RAW
    v16bf pa = load_A16x32(&Ps[wave][0][0], 32);

    // O = O*alpha + P @ V  (4 dim-chunks of 16)
#pragma unroll
    for (int t = 0; t < 4; ++t) {
      v8f ot = o[t];
#pragma unroll
      for (int r = 0; r < 8; ++r) ot[r] *= alpha[r];
      o[t] = wmma_bf16(pa, load_B32x16(&Vt[t * 16][0], 32), ot);
    }
  }

  const int hi = (lane & 16) ? 8 : 0;
  const int cn = lane & 15;
#pragma unroll
  for (int t = 0; t < 4; ++t)
#pragma unroll
    for (int r = 0; r < 8; ++r) {
      const int nrow = n0 + hi + r;
      const long orow = (long)b * oBatchStride + oRowOff + nrow;
      Op[orow * (long)Cc + hc + t * 16 + cn] = (OT)(o[t][r] / l_i[r]);
    }
}

// ---------------------------------------------------------------------------
// Pipeline
// ---------------------------------------------------------------------------
extern "C" void kernel_launch(void* const* d_in, const int* in_sizes, int n_in,
                              void* d_out, int out_size, void* d_ws, size_t ws_size,
                              hipStream_t stream) {
  const float* x           = (const float*)d_in[0];
  const float* w_qkv_diff  = (const float*)d_in[1];
  const float* w_qkv_cond  = (const float*)d_in[2];
  const float* w_proj_diff = (const float*)d_in[3];
  const float* b_proj_diff = (const float*)d_in[4];
  const float* w_proj_cond = (const float*)d_in[5];
  const float* b_proj_cond = (const float*)d_in[6];
  float* out = (float*)d_out;

  const int B = 4, N = 1024, C = 768, H = 12, C3 = 3 * C;
  const int M = B * N;
  const float scale_l2e = 0.125f * 1.44269504f;  // 64^-0.5 * log2(e)

  bf16* qkv_d   = (bf16*)d_ws;
  bf16* qkv_c   = qkv_d  + (size_t)M * C3;
  bf16* od_pre  = qkv_c  + (size_t)M * C3;
  bf16* oc_pre  = od_pre + (size_t)M * C;
  bf16* od_proj = oc_pre + (size_t)M * C;

  dim3 blk(256);
  dim3 gQKV(M / 128, C3 / 128);
  dim3 gP(M / 128, C / 128);
  dim3 gAtt(N / 128, H, B);

  // QKV projections (x is f32; x[:, :N] / x[:, N:] via batch-stride mapping)
  gemm_bf16_wmma<1024, float, bf16><<<gQKV, blk, 0, stream>>>(
      x, w_qkv_diff, nullptr, qkv_d, M, C3, C, 2048, 0,    N, 0);
  gemm_bf16_wmma<1024, float, bf16><<<gQKV, blk, 0, stream>>>(
      x, w_qkv_cond, nullptr, qkv_c, M, C3, C, 2048, 1024, N, 0);

  // attn_diff @ v_d ; attn_cond @ v_c
  flash_attn_wmma<bf16><<<gAtt, blk, 0, stream>>>(
      qkv_d, C3, qkv_d + C, C3, qkv_d + 2 * C, C3, od_pre, N, 0, N, N, C, scale_l2e);
  flash_attn_wmma<bf16><<<gAtt, blk, 0, stream>>>(
      qkv_c, C3, qkv_c + C, C3, qkv_c + 2 * C, C3, oc_pre, N, 0, N, N, C, scale_l2e);

  // o_d projection (+bias) — bf16 A path uses async-to-LDS staging
  gemm_bf16_wmma<1024, bf16, bf16><<<gP, blk, 0, stream>>>(
      od_pre, w_proj_diff, b_proj_diff, od_proj, M, C, C, N, 0, N, 0);

  // attn_cond applied to projected o_d -> first half of output (f32)
  flash_attn_wmma<float><<<gAtt, blk, 0, stream>>>(
      qkv_c, C3, qkv_c + C, C3, od_proj, C, out, 2 * N, 0, N, N, C, scale_l2e);

  // o_c projection (+bias) -> second half of output (f32)
  gemm_bf16_wmma<1024, bf16, float><<<gP, blk, 0, stream>>>(
      oc_pre, w_proj_cond, b_proj_cond, out, M, C, C, N, 0, 2 * N, 1024);
}
